// LongformerWithMotifsForSequenceClassification_89292370084510
// MI455X (gfx1250) — matrix operand order repaired and checked by reading, not compiled, not run on hardware
//
#include <hip/hip_runtime.h>
#include <cstdint>
#include <cstddef>

#define B_    2
#define S_    1024
#define L_    12
#define H_    12
#define D_    768
#define FF_   3072
#define DH_   64
#define MOTIF_ 414
#define NTOK  (B_ * S_)

typedef __attribute__((ext_vector_type(16))) _Float16 v16h;
typedef __attribute__((ext_vector_type(8)))  _Float16 v8h;
typedef __attribute__((ext_vector_type(4)))  _Float16 v4h;
typedef __attribute__((ext_vector_type(8)))  float    v8f;
typedef int v4i_b __attribute__((__vector_size__(4 * sizeof(int))));

#if defined(__has_builtin)
#if __has_builtin(__builtin_amdgcn_global_load_async_to_lds_b128)
#define ASYNC_LDS 1
#endif
#endif

// 16-byte global -> LDS copy; async DMA path on gfx1250 when available.
__device__ __forceinline__ void async_copy16(const float* gsrc, float* ldst) {
#ifdef ASYNC_LDS
  __builtin_amdgcn_global_load_async_to_lds_b128((v4i_b*)gsrc, (v4i_b*)ldst, 0, 0);
#else
  *(float4*)ldst = *(const float4*)gsrc;
#endif
}

__device__ __forceinline__ void async_wait() {
#ifdef ASYNC_LDS
  asm volatile("s_wait_asynccnt 0x0" ::: "memory");
#endif
}

// ---------------------------------------------------------------------------
// WMMA GEMM: Y[M,N] = A[M,K] * W[K,N] + bias   (f32 in/out, f16 WMMA core)
// Block: 256 threads (8 waves, 4Mx2N). Block tile 128(M) x 128(N), K step 32.
// Each wave: 32 rows x 64 cols = 2 A-frags x 4 B-frags = 8 WMMAs per K step.
// Double-buffered LDS pipeline: iteration i stores tile i+1 to buffer p^1,
// issues global loads for tile i+2, computes from buffer p, then one barrier.
// ---------------------------------------------------------------------------
#define TK 32
#define ASTR 40
#define BSTR 40

__launch_bounds__(256)
__global__ void gemm_wmma_kernel(const float* __restrict__ A,
                                 const float* __restrict__ W,
                                 const float* __restrict__ bias,
                                 float* __restrict__ Y,
                                 int M, int N, int K) {
  __shared__ __align__(16) _Float16 Asm[2][128 * ASTR];
  __shared__ __align__(16) _Float16 Bsm[2][128 * BSTR];

  const int tid  = threadIdx.x;
  const int lane = tid & 31;
  const int wave = tid >> 5;
  const int wm2  = wave >> 1;               // 0..3 : 32-row strip
  const int wn2  = wave & 1;                // 0..1 : 64-col strip
  const int m0   = blockIdx.x * 128;
  const int n0   = blockIdx.y * 128;

  float4 areg[4], breg[4];                  // staged next tile

  auto load_tile = [&](int k0) {
#pragma unroll
    for (int e = 0; e < 4; ++e) {           // A: 128x32 f32, 4 float4/thread
      const int i = e * 256 + tid;
      const int r = i >> 3, c4 = (i & 7) * 4;
      areg[e] = *(const float4*)&A[(size_t)(m0 + r) * K + k0 + c4];
    }
#pragma unroll
    for (int e = 0; e < 4; ++e) {           // B: 32x128 f32, 4 float4/thread
      const int i = e * 256 + tid;
      const int kr = i >> 5, c4 = (i & 31) * 4;
      breg[e] = *(const float4*)&W[(size_t)(k0 + kr) * N + n0 + c4];
    }
  };
  auto store_tile = [&](int buf) {
#pragma unroll
    for (int e = 0; e < 4; ++e) {
      const int i = e * 256 + tid;
      const int r = i >> 3, c4 = (i & 7) * 4;
      v4h h4 = { (_Float16)areg[e].x, (_Float16)areg[e].y,
                 (_Float16)areg[e].z, (_Float16)areg[e].w };
      *(v4h*)&Asm[buf][r * ASTR + c4] = h4;
    }
#pragma unroll
    for (int e = 0; e < 4; ++e) {           // B stored transposed: [n][k]
      const int i = e * 256 + tid;
      const int kr = i >> 5, c4 = (i & 31) * 4;
      Bsm[buf][(c4 + 0) * BSTR + kr] = (_Float16)breg[e].x;
      Bsm[buf][(c4 + 1) * BSTR + kr] = (_Float16)breg[e].y;
      Bsm[buf][(c4 + 2) * BSTR + kr] = (_Float16)breg[e].z;
      Bsm[buf][(c4 + 3) * BSTR + kr] = (_Float16)breg[e].w;
    }
  };

  v8f c00 = {}, c01 = {}, c02 = {}, c03 = {};
  v8f c10 = {}, c11 = {}, c12 = {}, c13 = {};

  // prologue: tile 0 into buffer 0, tile 1 staged in registers
  load_tile(0);
  store_tile(0);
  if (K > TK) load_tile(TK);
  __syncthreads();

  for (int k0 = 0; k0 < K; k0 += TK) {
    const int p = (k0 / TK) & 1;
    if (k0 + TK < K)     store_tile(p ^ 1);        // tile i+1 -> other buffer
    if (k0 + 2 * TK < K) load_tile(k0 + 2 * TK);   // tile i+2 -> registers

    // A fragments (ISA 7.12.2: 16-bit A 16x32, lanes 0-15 = rows)
    const int arow = wm2 * 32 + (lane & 15);
    const int koff = (lane >> 4) * 8;
    v8h a0lo = *(const v8h*)&Asm[p][arow * ASTR + koff];
    v8h a0hi = *(const v8h*)&Asm[p][arow * ASTR + koff + 16];
    v8h a1lo = *(const v8h*)&Asm[p][(arow + 16) * ASTR + koff];
    v8h a1hi = *(const v8h*)&Asm[p][(arow + 16) * ASTR + koff + 16];
    v16h af0 = __builtin_shufflevector(a0lo, a0hi,
        0, 1, 2, 3, 4, 5, 6, 7, 8, 9, 10, 11, 12, 13, 14, 15);
    v16h af1 = __builtin_shufflevector(a1lo, a1hi,
        0, 1, 2, 3, 4, 5, 6, 7, 8, 9, 10, 11, 12, 13, 14, 15);

    const int bn0 = wn2 * 64 + (lane & 15);
    const int bk  = (lane >> 4) * 8;
#pragma unroll
    for (int nt = 0; nt < 4; ++nt) {
      const int bn = bn0 + nt * 16;
      v8h blo = *(const v8h*)&Bsm[p][bn * BSTR + bk];
      v8h bhi = *(const v8h*)&Bsm[p][bn * BSTR + bk + 16];
      v16h bf = __builtin_shufflevector(blo, bhi,
          0, 1, 2, 3, 4, 5, 6, 7, 8, 9, 10, 11, 12, 13, 14, 15);
      v8f* cl = (nt == 0) ? &c00 : (nt == 1) ? &c01 : (nt == 2) ? &c02 : &c03;
      v8f* ch = (nt == 0) ? &c10 : (nt == 1) ? &c11 : (nt == 2) ? &c12 : &c13;
      *cl = __builtin_amdgcn_wmma_f32_16x16x32_f16(false, af0, false, bf,
                                                   (short)0, *cl, false, false);
      *ch = __builtin_amdgcn_wmma_f32_16x16x32_f16(false, af1, false, bf,
                                                   (short)0, *ch, false, false);
    }
    __syncthreads();
  }

  // epilogue: C layout = VGPR r -> row (r | r+8 by half-wave), N = lane&15
  const int crow = (lane >> 4) * 8;
  const int cn   = lane & 15;
  const int wmb  = m0 + wm2 * 32;
#pragma unroll
  for (int nt = 0; nt < 4; ++nt) {
    const int n = n0 + wn2 * 64 + nt * 16 + cn;
    const float bv = bias ? bias[n] : 0.0f;
    const v8f* cl = (nt == 0) ? &c00 : (nt == 1) ? &c01 : (nt == 2) ? &c02 : &c03;
    const v8f* ch = (nt == 0) ? &c10 : (nt == 1) ? &c11 : (nt == 2) ? &c12 : &c13;
#pragma unroll
    for (int r = 0; r < 8; ++r) {
      Y[(size_t)(wmb + crow + r) * N + n]      = (*cl)[r] + bv;
      Y[(size_t)(wmb + 16 + crow + r) * N + n] = (*ch)[r] + bv;
    }
  }
}

// ---------------------------------------------------------------------------
// Embeddings: tmp[b,s,d] = word[ids] + pos[s+2] + type
// ---------------------------------------------------------------------------
__global__ void embed_kernel(const int* __restrict__ ids,
                             const float* __restrict__ ew,
                             const float* __restrict__ ep,
                             const float* __restrict__ et,
                             float* __restrict__ Out) {
  const int gid = blockIdx.x * 256 + threadIdx.x;
  const int d = gid % D_;
  const int t = gid / D_;
  const int s = t & (S_ - 1);
  Out[gid] = ew[(size_t)ids[t] * D_ + d] + ep[(size_t)(s + 2) * D_ + d] + et[d];
}

// ---------------------------------------------------------------------------
// LayerNorm (optional fused residual): Out = LN(X + R). In-place safe.
// ---------------------------------------------------------------------------
__launch_bounds__(256)
__global__ void ln_kernel(const float* __restrict__ X, const float* __restrict__ R,
                          const float* __restrict__ gamma, const float* __restrict__ beta,
                          float* __restrict__ Out) {
  __shared__ float red[256];
  const int row = blockIdx.x;
  const int tid = threadIdx.x;
  const float* xp = X + (size_t)row * D_;
  float v0 = xp[tid], v1 = xp[tid + 256], v2 = xp[tid + 512];
  if (R) {
    const float* rp = R + (size_t)row * D_;
    v0 += rp[tid]; v1 += rp[tid + 256]; v2 += rp[tid + 512];
  }
  red[tid] = v0 + v1 + v2;
  __syncthreads();
  for (int st = 128; st > 0; st >>= 1) {
    if (tid < st) red[tid] += red[tid + st];
    __syncthreads();
  }
  const float mean = red[0] * (1.0f / 768.0f);
  __syncthreads();
  const float d0 = v0 - mean, d1 = v1 - mean, d2 = v2 - mean;
  red[tid] = d0 * d0 + d1 * d1 + d2 * d2;
  __syncthreads();
  for (int st = 128; st > 0; st >>= 1) {
    if (tid < st) red[tid] += red[tid + st];
    __syncthreads();
  }
  const float inv = rsqrtf(red[0] * (1.0f / 768.0f) + 1e-5f);
  float* op = Out + (size_t)row * D_;
  op[tid]       = d0 * inv * gamma[tid]       + beta[tid];
  op[tid + 256] = d1 * inv * gamma[tid + 256] + beta[tid + 256];
  op[tid + 512] = d2 * inv * gamma[tid + 512] + beta[tid + 512];
}

// ---------------------------------------------------------------------------
// Sliding-window attention. One block = 256 contiguous queries of one (b,h).
// K/V staged in 128-key LDS chunks (async-to-LDS when available); all lanes
// walk the chunk in lockstep -> LDS broadcast reads. Online softmax/thread.
// ---------------------------------------------------------------------------
__launch_bounds__(256)
__global__ void attn_local_kernel(const float* __restrict__ Q,
                                  const float* __restrict__ Kb,
                                  const float* __restrict__ Vb,
                                  const int* __restrict__ amask,
                                  float* __restrict__ Out) {
  __shared__ __align__(16) float Ks[128 * DH_];
  __shared__ __align__(16) float Vs[128 * DH_];
  __shared__ int Ms[128];

  const int bid = blockIdx.x;               // [0, B*H*4)
  const int s0  = (bid & 3) * 256;
  const int h   = (bid >> 2) % H_;
  const int b   = bid / (4 * H_);
  const int tid = threadIdx.x;
  const int s   = s0 + tid;
  const size_t bS = (size_t)b * S_;

  float qv[DH_];
  const float* qp = Q + (bS + s) * D_ + h * DH_;
#pragma unroll
  for (int d = 0; d < DH_; ++d) qv[d] = qp[d] * 0.125f;

  float m = -1e30f, lsum = 0.0f;
  float acc[DH_];
#pragma unroll
  for (int d = 0; d < DH_; ++d) acc[d] = 0.0f;

  const int lo = (s - 256 < 0) ? 0 : s - 256;
  const int hi = (s + 256 > S_ - 1) ? S_ - 1 : s + 256;

  // global CLS key (j = 0) when outside this query's window
  if (lo > 0 && amask[bS] != 0) {
    const float* kp = Kb + bS * D_ + h * DH_;
    float sc = 0.0f;
#pragma unroll
    for (int d = 0; d < DH_; ++d) sc += qv[d] * kp[d];
    m = sc; lsum = 1.0f;
    const float* vp = Vb + bS * D_ + h * DH_;
#pragma unroll
    for (int d = 0; d < DH_; ++d) acc[d] = vp[d];
  }

  for (int ci = 0; ci < 6; ++ci) {
    const int c0 = s0 - 256 + ci * 128;     // chunk key range [c0, c0+127]
    if (c0 > S_ - 1 || c0 + 127 < 0) continue;   // uniform across block
    __syncthreads();                        // previous chunk fully consumed
#pragma unroll
    for (int e = 0; e < 8; ++e) {           // 128 rows x 64 f32, x2 (K and V)
      const int seg = e * 256 + tid;
      const int r   = seg >> 4;
      const int c4  = (seg & 15) * 4;
      int j = c0 + r; j = j < 0 ? 0 : (j > S_ - 1 ? S_ - 1 : j);
      async_copy16(Kb + (bS + j) * D_ + h * DH_ + c4, &Ks[r * DH_ + c4]);
      async_copy16(Vb + (bS + j) * D_ + h * DH_ + c4, &Vs[r * DH_ + c4]);
    }
    if (tid < 128) {
      int j = c0 + tid; j = j < 0 ? 0 : (j > S_ - 1 ? S_ - 1 : j);
      Ms[tid] = amask[bS + j];
    }
    async_wait();
    __syncthreads();

    const int rlo = (lo - c0 < 0) ? 0 : lo - c0;
    const int rhi = (hi - c0 > 127) ? 127 : hi - c0;
    for (int r = rlo; r <= rhi; ++r) {
      if (Ms[r] == 0) continue;
      const float* kp = &Ks[r * DH_];
      float sc = 0.0f;
#pragma unroll
      for (int d = 0; d < DH_; ++d) sc += qv[d] * kp[d];
      const float mn = fmaxf(m, sc);
      const float corr = __expf(m - mn);
      const float p = __expf(sc - mn);
      lsum = lsum * corr + p;
      const float* vp = &Vs[r * DH_];
#pragma unroll
      for (int d = 0; d < DH_; ++d) acc[d] = acc[d] * corr + p * vp[d];
      m = mn;
    }
  }

  const float inv = (lsum > 0.0f) ? 1.0f / lsum : 0.0f;
  float* op = Out + (bS + s) * D_ + h * DH_;
#pragma unroll
  for (int d = 0; d < DH_; ++d) op[d] = acc[d] * inv;
}

// ---------------------------------------------------------------------------
// Global (CLS) attention: one block per (b,h); overwrites Out row s=0.
// ---------------------------------------------------------------------------
__launch_bounds__(256)
__global__ void attn_global_kernel(const float* __restrict__ qg,
                                   const float* __restrict__ KG,
                                   const float* __restrict__ VG,
                                   const int* __restrict__ amask,
                                   float* __restrict__ Out) {
  __shared__ float probs[S_];
  __shared__ float red[256];
  __shared__ float part[256];
  const int b = blockIdx.x / H_;
  const int h = blockIdx.x % H_;
  const int tid = threadIdx.x;
  const float* qp = qg + (size_t)b * D_ + h * DH_;

  float lmax = -1e30f;
  for (int j = tid; j < S_; j += 256) {
    const float* kp = KG + (size_t)(b * S_ + j) * D_ + h * DH_;
    float sc = 0.0f;
#pragma unroll
    for (int d = 0; d < DH_; ++d) sc += qp[d] * kp[d];
    sc *= 0.125f;
    if (amask[b * S_ + j] == 0) sc = -1e9f;
    probs[j] = sc;
    lmax = fmaxf(lmax, sc);
  }
  red[tid] = lmax;
  __syncthreads();
  for (int st = 128; st > 0; st >>= 1) {
    if (tid < st) red[tid] = fmaxf(red[tid], red[tid + st]);
    __syncthreads();
  }
  const float mx = red[0];
  __syncthreads();
  float ls = 0.0f;
  for (int j = tid; j < S_; j += 256) {
    const float p = __expf(probs[j] - mx);
    probs[j] = p;
    ls += p;
  }
  red[tid] = ls;
  __syncthreads();
  for (int st = 128; st > 0; st >>= 1) {
    if (tid < st) red[tid] += red[tid + st];
    __syncthreads();
  }
  const float inv = 1.0f / red[0];
  __syncthreads();

  const int d = tid & 63;
  const int chunk = tid >> 6;
  float a = 0.0f;
  for (int j = chunk * 256; j < chunk * 256 + 256; ++j)
    a += probs[j] * VG[(size_t)(b * S_ + j) * D_ + h * DH_ + d];
  part[tid] = a;
  __syncthreads();
  if (tid < 64) {
    const float r = (part[tid] + part[tid + 64] + part[tid + 128] + part[tid + 192]) * inv;
    Out[(size_t)(b * S_) * D_ + h * DH_ + tid] = r;
  }
}

// ---------------------------------------------------------------------------
__global__ void gelu_kernel(float* __restrict__ X) {
  const size_t gid = (size_t)blockIdx.x * 256 + threadIdx.x;
  const float v = X[gid];
  X[gid] = 0.5f * v * (1.0f + erff(v * 0.70710678118654752f));
}

__global__ void qg_kernel(const float* __restrict__ X, const float* __restrict__ W,
                          const float* __restrict__ bias, float* __restrict__ Out) {
  const int gid = blockIdx.x * 256 + threadIdx.x;
  const int n = gid % D_;
  const int b = gid / D_;
  const float* xp = X + (size_t)(b * S_) * D_;
  float s = bias[n];
  for (int k = 0; k < D_; ++k) s += xp[k] * W[(size_t)k * D_ + n];
  Out[b * D_ + n] = s;
}

__global__ void head1_kernel(const float* __restrict__ X, const float* __restrict__ motif,
                             const float* __restrict__ Wd, const float* __restrict__ bd,
                             float* __restrict__ hd) {
  const int gid = blockIdx.x * 256 + threadIdx.x;
  const int n = gid % D_;
  const int b = gid / D_;
  float s = bd[n];
  for (int k = 0; k < D_ + MOTIF_; ++k) {
    const float hv = (k < D_) ? X[(size_t)(b * S_) * D_ + k] : motif[b * MOTIF_ + (k - D_)];
    s += hv * Wd[(size_t)k * D_ + n];
  }
  hd[b * D_ + n] = tanhf(s);
}

__global__ void head2_kernel(const float* __restrict__ hd, const float* __restrict__ Wp,
                             const float* __restrict__ bp, float* __restrict__ out) {
  const int tid = threadIdx.x;
  if (tid >= B_ * 2) return;
  const int b = tid >> 1;
  const int c = tid & 1;
  float s = bp[c];
  for (int n = 0; n < D_; ++n) s += hd[b * D_ + n] * Wp[n * 2 + c];
  out[b * 2 + c] = s;
}

// ---------------------------------------------------------------------------
extern "C" void kernel_launch(void* const* d_in, const int* in_sizes, int n_in,
                              void* d_out, int out_size, void* d_ws, size_t ws_size,
                              hipStream_t stream) {
  (void)in_sizes; (void)n_in; (void)out_size; (void)ws_size;
  const int*   ids      = (const int*)  d_in[0];
  const int*   amask    = (const int*)  d_in[1];
  const float* motif    = (const float*)d_in[2];
  const float* emb_word = (const float*)d_in[3];
  const float* emb_pos  = (const float*)d_in[4];
  const float* emb_type = (const float*)d_in[5];
  const float* emb_ln_s = (const float*)d_in[6];
  const float* emb_ln_b = (const float*)d_in[7];
  const float* Wq  = (const float*)d_in[8];   const float* bq  = (const float*)d_in[9];
  const float* Wk  = (const float*)d_in[10];  const float* bk  = (const float*)d_in[11];
  const float* Wv  = (const float*)d_in[12];  const float* bv  = (const float*)d_in[13];
  const float* Wqg = (const float*)d_in[14];  const float* bqg = (const float*)d_in[15];
  const float* Wkg = (const float*)d_in[16];  const float* bkg = (const float*)d_in[17];
  const float* Wvg = (const float*)d_in[18];  const float* bvg = (const float*)d_in[19];
  const float* Wo  = (const float*)d_in[20];  const float* bo  = (const float*)d_in[21];
  const float* ln1s= (const float*)d_in[22];  const float* ln1b= (const float*)d_in[23];
  const float* Wi  = (const float*)d_in[24];  const float* bi  = (const float*)d_in[25];
  const float* Wo2 = (const float*)d_in[26];  const float* bo2 = (const float*)d_in[27];
  const float* ln2s= (const float*)d_in[28];  const float* ln2b= (const float*)d_in[29];
  const float* hWd = (const float*)d_in[30];  const float* hbd = (const float*)d_in[31];
  const float* hWp = (const float*)d_in[32];  const float* hbp = (const float*)d_in[33];

  float* ws = (float*)d_ws;
  const size_t XSZ = (size_t)NTOK * D_;
  float* x    = ws + 0 * XSZ;
  float* tmp  = ws + 1 * XSZ;
  float* aout = ws + 2 * XSZ;
  float* qb   = ws + 3 * XSZ;
  float* kb   = ws + 4 * XSZ;
  float* vb   = ws + 5 * XSZ;
  float* kgb  = ws + 6 * XSZ;
  float* vgb  = ws + 7 * XSZ;
  float* hbuf = ws + 3 * XSZ;               // FFN buffer aliases qb..kgb
  float* qgb  = ws + 8 * XSZ;
  float* hd   = ws + 8 * XSZ + 2048;

  const dim3 blk(256);
  const dim3 ggrid(NTOK / 128, D_ / 128);   // (16, 6)
  const dim3 ggrid_ff(NTOK / 128, FF_ / 128); // (16, 24)

  embed_kernel<<<dim3((NTOK * D_) / 256), blk, 0, stream>>>(ids, emb_word, emb_pos,
                                                            emb_type, tmp);
  ln_kernel<<<dim3(NTOK), blk, 0, stream>>>(tmp, nullptr, emb_ln_s, emb_ln_b, x);

  for (int l = 0; l < L_; ++l) {
    const size_t wofs = (size_t)l * D_ * D_;
    const size_t bofs = (size_t)l * D_;
    gemm_wmma_kernel<<<ggrid, blk, 0, stream>>>(x, Wq + wofs, bq + bofs, qb, NTOK, D_, D_);
    gemm_wmma_kernel<<<ggrid, blk, 0, stream>>>(x, Wk + wofs, bk + bofs, kb, NTOK, D_, D_);
    gemm_wmma_kernel<<<ggrid, blk, 0, stream>>>(x, Wv + wofs, bv + bofs, vb, NTOK, D_, D_);
    gemm_wmma_kernel<<<ggrid, blk, 0, stream>>>(x, Wkg + wofs, bkg + bofs, kgb, NTOK, D_, D_);
    gemm_wmma_kernel<<<ggrid, blk, 0, stream>>>(x, Wvg + wofs, bvg + bofs, vgb, NTOK, D_, D_);
    qg_kernel<<<dim3((B_ * D_) / 256), blk, 0, stream>>>(x, Wqg + wofs, bqg + bofs, qgb);
    attn_local_kernel<<<dim3(B_ * H_ * 4), blk, 0, stream>>>(qb, kb, vb, amask, aout);
    attn_global_kernel<<<dim3(B_ * H_), blk, 0, stream>>>(qgb, kgb, vgb, amask, aout);
    gemm_wmma_kernel<<<ggrid, blk, 0, stream>>>(aout, Wo + wofs, bo + bofs, tmp, NTOK, D_, D_);
    ln_kernel<<<dim3(NTOK), blk, 0, stream>>>(x, tmp, ln1s + bofs, ln1b + bofs, x);
    gemm_wmma_kernel<<<ggrid_ff, blk, 0, stream>>>(x, Wi + (size_t)l * D_ * FF_,
                                                   bi + (size_t)l * FF_, hbuf, NTOK, FF_, D_);
    gelu_kernel<<<dim3((unsigned)(((size_t)NTOK * FF_) / 256)), blk, 0, stream>>>(hbuf);
    gemm_wmma_kernel<<<ggrid, blk, 0, stream>>>(hbuf, Wo2 + (size_t)l * FF_ * D_,
                                                bo2 + bofs, tmp, NTOK, D_, FF_);
    ln_kernel<<<dim3(NTOK), blk, 0, stream>>>(x, tmp, ln2s + bofs, ln2b + bofs, x);
  }

  head1_kernel<<<dim3((B_ * D_) / 256), blk, 0, stream>>>(x, motif, hWd, hbd, hd);
  head2_kernel<<<dim3(1), dim3(64), 0, stream>>>(hd, hWp, hbp, (float*)d_out);
}
